// DeformAttn_32753420599766
// MI455X (gfx1250) — matrix operand order repaired and checked by reading, not compiled, not run on hardware
//
#include <hip/hip_runtime.h>
#include <hip/hip_bf16.h>
#include <math.h>

// ---------------------------------------------------------------------------
// DeformAttn for MI455X (gfx1250, wave32, WMMA).
// B=4, C=64, H=W=64. Flash-attention streaming softmax avoids materializing
// the 4x4096x4096 score matrix (256MB -> ~44us of HBM at 23.3TB/s).
// GEMMs use v_wmma_f32_16x16x32_f16; K/V chunks stream through LDS via the
// CDNA5 async global->LDS path (global_load_async_to_lds_b128 + ASYNCcnt).
// ---------------------------------------------------------------------------

typedef __attribute__((ext_vector_type(16))) _Float16 v16h;
typedef __attribute__((ext_vector_type(8)))  float    v8f;

constexpr int Bn  = 4;
constexpr int Cc  = 64;
constexpr int Hh  = 64;
constexpr int Ww  = 64;
constexpr int HW  = Hh * Ww;           // 4096
constexpr float SCALE = 0.125f;        // 64^-0.5
constexpr float OFR   = 4.0f;
constexpr int NCHUNK = 32;             // key/value streaming chunk

// --------------------------- WMMA fragment loaders -------------------------
// Layouts used here (wave32; contiguous K, not the ISA table order):
// A 16x32 f16 : lane<16 -> row M=lane, K in {0..15}; lane>=16 -> M=lane-16,
//               K in {16..31}. half h: K = (lane<16?0:16) + h.
// B 32x16 f16 : lane -> col N=lane&15, K = (lane<16?0:16) + h.
// C/D 16x16 f32: float f -> row M=(lane<16? f : f+8), col N=lane&15.

__device__ __forceinline__ v16h load_a_rm(const _Float16* __restrict__ base,
                                          int ld, int lane) {
  // element (M,K) at base[M*ld + K]
  v16h a;
  const _Float16* p = base + (size_t)(lane & 15) * ld;
  const int kb0 = (lane < 16) ? 0 : 16;
  const int kb1 = (lane < 16) ? 8 : 24;
#pragma unroll
  for (int i = 0; i < 8; ++i) {
    a[i]     = p[kb0 + i];
    a[8 + i] = p[kb1 + i];
  }
  return a;
}

__device__ __forceinline__ v16h load_a_cw_f32(const float* __restrict__ base,
                                              int ldK, int lane) {
  // element (M,K) at base[K*ldK + M]  (channel-major f32 source)
  v16h a;
  const int m   = lane & 15;
  const int kb0 = (lane < 16) ? 0 : 16;
  const int kb1 = (lane < 16) ? 8 : 24;
#pragma unroll
  for (int i = 0; i < 8; ++i) {
    a[i]     = (_Float16)base[(size_t)(kb0 + i) * ldK + m];
    a[8 + i] = (_Float16)base[(size_t)(kb1 + i) * ldK + m];
  }
  return a;
}

__device__ __forceinline__ v16h load_b_col_f16(const _Float16* base,
                                               int ld, int lane) {
  // element (K,N) at base[N*ld + K]   (K contiguous per column)
  v16h r;
  const _Float16* p = base + (size_t)(lane & 15) * ld + ((lane < 16) ? 0 : 16);
#pragma unroll
  for (int h = 0; h < 16; ++h) r[h] = p[h];
  return r;
}

__device__ __forceinline__ v16h load_b_col_f32(const float* __restrict__ base,
                                               int ld, int lane) {
  // element (K,N) at base[N*ld + K], f32 source converted to f16
  v16h r;
  const float* p = base + (size_t)(lane & 15) * ld + ((lane < 16) ? 0 : 16);
#pragma unroll
  for (int h = 0; h < 16; ++h) r[h] = (_Float16)p[h];
  return r;
}

__device__ __forceinline__ v16h load_b_row_f16(const _Float16* base,
                                               int ld, int lane) {
  // element (K,N) at base[K*ld + N]   (row-major in K)
  v16h r;
  const _Float16* p = base + (lane & 15) + (size_t)((lane < 16) ? 0 : 16) * ld;
#pragma unroll
  for (int h = 0; h < 16; ++h) r[h] = p[(size_t)h * ld];
  return r;
}

__device__ __forceinline__ v8f wmma_f16(v16h a, v16h b, v8f c) {
  return __builtin_amdgcn_wmma_f32_16x16x32_f16(false, a, false, b,
                                                (short)0, c, false, false);
}

// ------------------- CDNA5 async global->LDS (ASYNCcnt) ---------------------
// Inline asm: portable across ROCm7.2 / amdgpu-toolchain builtin differences.
__device__ __forceinline__ void async_copy_b128(const _Float16* g, _Float16* l) {
  typedef __attribute__((address_space(3))) _Float16 lds_h;
  const unsigned loff = (unsigned)(unsigned long long)(lds_h*)l;  // LDS byte addr
  asm volatile("global_load_async_to_lds_b128 %0, %1, off"
               :: "v"(loff), "v"(g)
               : "memory");
}
__device__ __forceinline__ void wait_async_le(int) = delete;
__device__ __forceinline__ void wait_async_le4() {
  asm volatile("s_wait_asynccnt 0x4" ::: "memory");
}
__device__ __forceinline__ void wait_async_0() {
  asm volatile("s_wait_asynccnt 0x0" ::: "memory");
}

// --------------------- Kernel 1: q = wq * prompt + bq -----------------------
// A from channel-major f32, outputs: f16 pixel-major (attn) + f32 ch-major (dwconv)
__global__ __launch_bounds__(256) void gemm_cw_kernel(
    const float* __restrict__ X,     // [B][C][HW]
    const float* __restrict__ Wt,    // [O][C]
    const float* __restrict__ bias,  // [O]
    _Float16* __restrict__ Y16,      // [B][HW][O]
    float* __restrict__ Y32)         // [B][O][HW]
{
  const int lane = threadIdx.x & 31;
  const int wave = threadIdx.x >> 5;
  const int tile = blockIdx.x * 8 + wave;       // Bn * (HW/16) * (Cc/16) tiles
  const int ntn  = Cc / 16;                     // 4
  const int ntm  = HW / 16;                     // 256
  const int b  = tile / (ntm * ntn);
  const int r  = tile % (ntm * ntn);
  const int mt = r / ntn;
  const int nt = r % ntn;

  const float* Xb = X + (size_t)b * Cc * HW + mt * 16;
  v8f acc = {};
#pragma unroll
  for (int ks = 0; ks < 2; ++ks) {
    v16h a = load_a_cw_f32(Xb + (size_t)(ks * 32) * HW, HW, lane);
    v16h w = load_b_col_f32(Wt + (size_t)(nt * 16) * Cc + ks * 32, Cc, lane);
    acc = wmma_f16(a, w, acc);
  }
  const int col = lane & 15;
  const float bv = bias[nt * 16 + col];
#pragma unroll
  for (int f = 0; f < 8; ++f) {
    const int row = (lane < 16) ? f : (f + 8);
    const float val = acc[f] + bv;
    Y16[((size_t)b * HW + mt * 16 + row) * Cc + nt * 16 + col] = (_Float16)val;
    Y32[((size_t)b * Cc + nt * 16 + col) * HW + mt * 16 + row] = val;
  }
}

// ------ Kernel 2: dwconv3x3 + LN + GELU + offset + tanh + bilinear sample ---
// One wave per pixel; lane owns channels (lane, lane+32).
__global__ __launch_bounds__(256) void fused_pixel_kernel(
    const float* __restrict__ q32,   // [B][C][HW]
    const float* __restrict__ kv,    // [B][C][HW]
    const float* __restrict__ dw_w,  // [C][9]
    const float* __restrict__ dw_b,  // [C]
    const float* __restrict__ ln_w,  // [C]
    const float* __restrict__ ln_b,  // [C]
    const float* __restrict__ off_w, // [2][C]
    _Float16* __restrict__ xs16)     // [B][HW][C]
{
  const int lane = threadIdx.x & 31;
  const int wave = threadIdx.x >> 5;
  const int pix  = blockIdx.x * 8 + wave;   // 0 .. B*HW-1
  const int b  = pix / HW;
  const int hw = pix % HW;
  const int y  = hw / Ww;
  const int x  = hw % Ww;

  // depthwise 3x3, zero padded
  float t[2];
#pragma unroll
  for (int i = 0; i < 2; ++i) {
    const int c = lane + 32 * i;
    const float* qc = q32 + ((size_t)b * Cc + c) * HW;
    const float* wc = dw_w + c * 9;
    float acc = dw_b[c];
#pragma unroll
    for (int ky = 0; ky < 3; ++ky) {
      const int yy = y + ky - 1;
      if (yy < 0 || yy >= Hh) continue;
#pragma unroll
      for (int kx = 0; kx < 3; ++kx) {
        const int xx = x + kx - 1;
        if (xx < 0 || xx >= Ww) continue;
        acc += qc[yy * Ww + xx] * wc[ky * 3 + kx];
      }
    }
    t[i] = acc;
  }

  // LayerNorm over 64 channels (wave32 reduction, 2 ch/lane)
  float s1 = t[0] + t[1];
  float s2 = t[0] * t[0] + t[1] * t[1];
#pragma unroll
  for (int m = 1; m < 32; m <<= 1) {
    s1 += __shfl_xor(s1, m, 32);
    s2 += __shfl_xor(s2, m, 32);
  }
  const float mean = s1 * (1.0f / 64.0f);
  const float var  = s2 * (1.0f / 64.0f) - mean * mean;
  const float rstd = rsqrtf(var + 1e-5f);

  // GELU (exact) + offset projection (2 x C dot products)
  float po0 = 0.f, po1 = 0.f;
#pragma unroll
  for (int i = 0; i < 2; ++i) {
    const int c = lane + 32 * i;
    const float u  = (t[i] - mean) * rstd * ln_w[c] + ln_b[c];
    const float ge = 0.5f * u * (1.0f + erff(u * 0.70710678118f));
    po0 += off_w[c] * ge;        // offset channel 0 (-> y)
    po1 += off_w[Cc + c] * ge;   // offset channel 1 (-> x)
  }
#pragma unroll
  for (int m = 1; m < 32; m <<= 1) {
    po0 += __shfl_xor(po0, m, 32);
    po1 += __shfl_xor(po1, m, 32);
  }

  // tanh-bounded offset + reference grid -> sampling position (wave-uniform)
  const float oy = tanhf(po0) * (OFR / (Hh - 1));
  const float ox = tanhf(po1) * (OFR / (Ww - 1));
  const float ry = ((float)y + 0.5f) / (Hh - 1.0f) * 2.0f - 1.0f;
  const float rx = ((float)x + 0.5f) / (Ww - 1.0f) * 2.0f - 1.0f;
  const float gx = ((ox + rx) + 1.0f) * 0.5f * (Ww - 1);
  const float gy = ((oy + ry) + 1.0f) * 0.5f * (Hh - 1);

  const float x0f = floorf(gx), y0f = floorf(gy);
  const float wx = gx - x0f, wy = gy - y0f;
  const int ix0 = (int)x0f, iy0 = (int)y0f;
  const int ix1 = ix0 + 1,  iy1 = iy0 + 1;
  const float mx0 = (ix0 >= 0 && ix0 < Ww) ? 1.f : 0.f;
  const float mx1 = (ix1 >= 0 && ix1 < Ww) ? 1.f : 0.f;
  const float my0 = (iy0 >= 0 && iy0 < Hh) ? 1.f : 0.f;
  const float my1 = (iy1 >= 0 && iy1 < Hh) ? 1.f : 0.f;
  const int cx0 = min(max(ix0, 0), Ww - 1);
  const int cx1 = min(max(ix1, 0), Ww - 1);
  const int cy0 = min(max(iy0, 0), Hh - 1);
  const int cy1 = min(max(iy1, 0), Hh - 1);
  const float w00 = mx0 * my0 * (1.f - wx) * (1.f - wy);
  const float w01 = mx1 * my0 * wx * (1.f - wy);
  const float w10 = mx0 * my1 * (1.f - wx) * wy;
  const float w11 = mx1 * my1 * wx * wy;

  const float* kvb = kv + (size_t)b * Cc * HW;
#pragma unroll
  for (int i = 0; i < 2; ++i) {
    const int c = lane + 32 * i;
    const float* kc = kvb + (size_t)c * HW;
    const float val = kc[cy0 * Ww + cx0] * w00 + kc[cy0 * Ww + cx1] * w01 +
                      kc[cy1 * Ww + cx0] * w10 + kc[cy1 * Ww + cx1] * w11;
    xs16[((size_t)b * HW + hw) * Cc + c] = (_Float16)val;
  }
}

// ---------------- Kernel 3/4/6: GEMM from pixel-major f16 -------------------
__global__ __launch_bounds__(256) void gemm_pm_kernel(
    const _Float16* __restrict__ A,  // [B][HW][C]
    const float* __restrict__ Wt,    // [O][C]
    const float* __restrict__ bias,  // [O]
    _Float16* __restrict__ Y16,      // [B][HW][O]  (or null)
    float* __restrict__ Y32)         // [B][O][HW]  (or null)
{
  const int lane = threadIdx.x & 31;
  const int wave = threadIdx.x >> 5;
  const int tile = blockIdx.x * 8 + wave;
  const int ntn  = Cc / 16;
  const int ntm  = HW / 16;
  const int b  = tile / (ntm * ntn);
  const int r  = tile % (ntm * ntn);
  const int mt = r / ntn;
  const int nt = r % ntn;

  const _Float16* Ab = A + ((size_t)b * HW + mt * 16) * Cc;
  v8f acc = {};
#pragma unroll
  for (int ks = 0; ks < 2; ++ks) {
    v16h a = load_a_rm(Ab + ks * 32, Cc, lane);
    v16h w = load_b_col_f32(Wt + (size_t)(nt * 16) * Cc + ks * 32, Cc, lane);
    acc = wmma_f16(a, w, acc);
  }
  const int col = lane & 15;
  const float bv = bias[nt * 16 + col];
#pragma unroll
  for (int f = 0; f < 8; ++f) {
    const int row = (lane < 16) ? f : (f + 8);
    const float val = acc[f] + bv;
    if (Y16) Y16[((size_t)b * HW + mt * 16 + row) * Cc + nt * 16 + col] = (_Float16)val;
    if (Y32) Y32[((size_t)b * Cc + nt * 16 + col) * HW + mt * 16 + row] = val;
  }
}

// --------------------- Kernel 5: flash attention ----------------------------
// 4 waves/block; each wave owns 16 query rows. K/V chunks are staged into LDS
// once per block via async global->LDS b128 copies (double buffered, ASYNCcnt),
// overlapping the next chunk's DMA with this chunk's WMMA + online softmax.
__global__ __launch_bounds__(128) void flash_kernel(
    const _Float16* __restrict__ q16,  // [B][M][C]
    const _Float16* __restrict__ k16,  // [B][N][C]
    const _Float16* __restrict__ v16,  // [B][N][C]
    _Float16* __restrict__ o16)        // [B][M][C]
{
  __shared__ _Float16 ldsK[2][NCHUNK][Cc];   // 8 KB double-buffered K chunk
  __shared__ _Float16 ldsV[2][NCHUNK][Cc];   // 8 KB double-buffered V chunk
  __shared__ _Float16 ldsP[4][16][NCHUNK];   // per-wave P staging (C->A layout)

  const int tid  = threadIdx.x;
  const int lane = tid & 31;
  const int wave = tid >> 5;
  const int b      = blockIdx.x / (HW / 64);
  const int mblock = blockIdx.x % (HW / 64);
  const int m0     = mblock * 64 + wave * 16;

  const _Float16* Qp = q16 + ((size_t)b * HW + m0) * Cc;
  const _Float16* Kp = k16 + (size_t)b * HW * Cc;
  const _Float16* Vp = v16 + (size_t)b * HW * Cc;

  const v16h qa0 = load_a_rm(Qp, Cc, lane);        // K = 0..31
  const v16h qa1 = load_a_rm(Qp + 32, Cc, lane);   // K = 32..63

  v8f o[4] = {v8f{}, v8f{}, v8f{}, v8f{}};
  float rmax[8], rsum[8];
#pragma unroll
  for (int f = 0; f < 8; ++f) { rmax[f] = -INFINITY; rsum[f] = 0.f; }

  // stage chunk into buffer: 32 rows x 128B = 256 b128 segments; 2 per thread
  auto stage = [&](const _Float16* gK, const _Float16* gV, int buf) {
#pragma unroll
    for (int i = 0; i < 2; ++i) {
      const int seg = tid + i * 128;          // 0..255
      const int row = seg >> 3;
      const int off = (seg & 7) * 8;          // 8 halfs = 16B
      async_copy_b128(gK + (size_t)row * Cc + off, &ldsK[buf][row][off]);
      async_copy_b128(gV + (size_t)row * Cc + off, &ldsV[buf][row][off]);
    }
  };

  stage(Kp, Vp, 0);                            // prologue: chunk 0

  for (int it = 0; it < HW / NCHUNK; ++it) {
    const int n0  = it * NCHUNK;
    const int cur = it & 1;
    const bool more = (n0 + NCHUNK) < HW;
    if (more)
      stage(Kp + (size_t)(n0 + NCHUNK) * Cc, Vp + (size_t)(n0 + NCHUNK) * Cc,
            cur ^ 1);
    // current chunk's 4 async ops done; next chunk's 4 may still be in flight
    if (more) wait_async_le4(); else wait_async_0();
    __syncthreads();

    // S = scale * Q K^T for this 16m x 32n chunk (two 16x16 subtiles)
    v8f s[2];
#pragma unroll
    for (int j = 0; j < 2; ++j) {
      const _Float16* kb = &ldsK[cur][16 * j][0];
      v16h b0 = load_b_col_f16(kb, Cc, lane);        // channels 0..31
      v16h b1 = load_b_col_f16(kb + 32, Cc, lane);   // channels 32..63
      v8f t = {};
      t = wmma_f16(qa0, b0, t);
      t = wmma_f16(qa1, b1, t);
#pragma unroll
      for (int f = 0; f < 8; ++f) t[f] *= SCALE;
      s[j] = t;
    }

    // online softmax: per-row max over the chunk (reduce across 16-lane half)
    float nmax[8];
#pragma unroll
    for (int f = 0; f < 8; ++f) nmax[f] = fmaxf(s[0][f], s[1][f]);
#pragma unroll
    for (int m = 1; m < 16; m <<= 1)
#pragma unroll
      for (int f = 0; f < 8; ++f) nmax[f] = fmaxf(nmax[f], __shfl_xor(nmax[f], m, 32));
#pragma unroll
    for (int f = 0; f < 8; ++f) nmax[f] = fmaxf(nmax[f], rmax[f]);

    float alpha[8], psum[8];
#pragma unroll
    for (int f = 0; f < 8; ++f) { alpha[f] = __expf(rmax[f] - nmax[f]); psum[f] = 0.f; }

    // P = exp(S - max) -> LDS (convert C-fragment layout to A-fragment layout)
#pragma unroll
    for (int j = 0; j < 2; ++j)
#pragma unroll
      for (int f = 0; f < 8; ++f) {
        const float p = __expf(s[j][f] - nmax[f]);
        psum[f] += p;
        const int row = (lane < 16) ? f : (f + 8);
        ldsP[wave][row][16 * j + (lane & 15)] = (_Float16)p;
      }
#pragma unroll
    for (int m = 1; m < 16; m <<= 1)
#pragma unroll
      for (int f = 0; f < 8; ++f) psum[f] += __shfl_xor(psum[f], m, 32);

#pragma unroll
    for (int f = 0; f < 8; ++f) {
      rsum[f] = rsum[f] * alpha[f] + psum[f];
      rmax[f] = nmax[f];
    }
#pragma unroll
    for (int t = 0; t < 4; ++t)
#pragma unroll
      for (int f = 0; f < 8; ++f) o[t][f] *= alpha[f];

    __syncthreads();   // P visible across lanes before A-layout reload

    const v16h pa = load_a_rm(&ldsP[wave][0][0], NCHUNK, lane);  // 16m x 32k
#pragma unroll
    for (int t = 0; t < 4; ++t) {
      v16h vb = load_b_row_f16(&ldsV[cur][0][t * 16], Cc, lane);
      o[t] = wmma_f16(pa, vb, o[t]);
    }
    __syncthreads();   // all reads of ldsK/ldsV[cur] done before restaging it
  }

  // normalize and store f16 pixel-major
  const int col = lane & 15;
#pragma unroll
  for (int t = 0; t < 4; ++t)
#pragma unroll
    for (int f = 0; f < 8; ++f) {
      const int row = (lane < 16) ? f : (f + 8);
      o16[((size_t)b * HW + m0 + row) * Cc + t * 16 + col] =
          (_Float16)(o[t][f] / rsum[f]);
    }
}

// ---------------------------------------------------------------------------
extern "C" void kernel_launch(void* const* d_in, const int* in_sizes, int n_in,
                              void* d_out, int out_size, void* d_ws, size_t ws_size,
                              hipStream_t stream) {
  const float* prompt = (const float*)d_in[0];
  const float* kv     = (const float*)d_in[1];
  const float* wq     = (const float*)d_in[2];
  const float* bq     = (const float*)d_in[3];
  const float* wk     = (const float*)d_in[4];
  const float* bk     = (const float*)d_in[5];
  const float* wv     = (const float*)d_in[6];
  const float* bv     = (const float*)d_in[7];
  const float* wo     = (const float*)d_in[8];
  const float* bo     = (const float*)d_in[9];
  const float* dw_w   = (const float*)d_in[10];
  const float* dw_b   = (const float*)d_in[11];
  const float* ln_w   = (const float*)d_in[12];
  const float* ln_b   = (const float*)d_in[13];
  const float* off_w  = (const float*)d_in[14];

  // workspace: 14 MiB total
  char* ws = (char*)d_ws;
  float*     q32   = (float*)ws;                          // 4 MiB [B][C][HW]
  _Float16*  q16   = (_Float16*)(ws + (4ull  << 20));     // 2 MiB [B][HW][C]
  _Float16*  xs16  = (_Float16*)(ws + (6ull  << 20));     // 2 MiB [B][HW][C]
  _Float16*  k16   = (_Float16*)(ws + (8ull  << 20));     // 2 MiB [B][HW][C]
  _Float16*  v16   = (_Float16*)(ws + (10ull << 20));     // 2 MiB [B][HW][C]
  _Float16*  a16   = (_Float16*)(ws + (12ull << 20));     // 2 MiB [B][HW][C]

  // 1. q projection (WMMA): f32 ch-major for dwconv + f16 pixel-major for attn
  gemm_cw_kernel<<<512, 256, 0, stream>>>(prompt, wq, bq, q16, q32);
  // 2. dwconv + LN + GELU + offset + tanh + bilinear sample -> xs (f16)
  fused_pixel_kernel<<<2048, 256, 0, stream>>>(q32, kv, dw_w, dw_b, ln_w, ln_b,
                                               off_w, xs16);
  // 3/4. k,v projections (WMMA)
  gemm_pm_kernel<<<512, 256, 0, stream>>>(xs16, wk, bk, k16, nullptr);
  gemm_pm_kernel<<<512, 256, 0, stream>>>(xs16, wv, bv, v16, nullptr);
  // 5. flash attention (async LDS K/V staging + WMMA QK^T / PV + online softmax)
  flash_kernel<<<256, 128, 0, stream>>>(q16, k16, v16, a16);
  // 6. output projection -> fp32 channel-major d_out
  gemm_pm_kernel<<<512, 256, 0, stream>>>(a16, wo, bo, nullptr, (float*)d_out);
}
